// SparseLocalityEnhancedMLP_4904852652263
// MI455X (gfx1250) — matrix operand order repaired and checked by reading, not compiled, not run on hardware
//
#include <hip/hip_runtime.h>
#include <hip/hip_bf16.h>
#include <math.h>
#include <stdint.h>

// Problem constants (match reference)
#define B_   4
#define K_   4096
#define C_   384
#define HID_ 1536
#define T_   8
#define H_   32
#define W_   32
#define L_   (T_ * H_ * W_)   // 8192

typedef __attribute__((ext_vector_type(16))) __bf16 v16bf;
typedef __attribute__((ext_vector_type(8)))  __bf16 v8bf;
typedef __attribute__((ext_vector_type(8)))  float  v8f;
typedef __attribute__((ext_vector_type(4)))  unsigned int v4u;
typedef __attribute__((ext_vector_type(8)))  int v8i;
typedef __attribute__((ext_vector_type(4)))  int v4i;

// A fragment K-base in this code's contiguous order (k = 16*half + 2v): VGPR pair v, lane half
__device__ __forceinline__ int kbaseA(int v, int half) {
    return (v << 1) + (half << 4);
}

// ---- TDM: DMA a tile_y x tile_x panel (elements of size 2^dsize_code bytes,
// row-major, row stride stride_x elements) from global into LDS at lds_off.
// D# per ISA §8. 6-arg builtin on this toolchain; groups 2/3 zero for 2-D.
__device__ __forceinline__ void tdm_load_tile(unsigned lds_off, const void* gptr,
                                              int dsize_code,
                                              int tile_x, int tile_y,
                                              int tensor_x, int tensor_y,
                                              int stride_x) {
    unsigned long long ga = (unsigned long long)(uintptr_t)gptr;
    v4u g0;
    g0[0] = 1u;                                                 // count=1 (valid user D#)
    g0[1] = lds_off;                                            // lds_addr (bytes)
    g0[2] = (unsigned)ga;                                       // global_addr[31:0]
    g0[3] = (unsigned)((ga >> 32) & 0x01ffffffu) | (2u << 30);  // addr[56:32] | type=2
    v8i g1;
    g1[0] = (int)((unsigned)dsize_code << 16);                  // data_size
    g1[1] = (int)(((unsigned)tensor_x & 0xffffu) << 16);        // tensor_dim0[15:0]
    g1[2] = (int)((((unsigned)tensor_x >> 16) & 0xffffu) |
                  (((unsigned)tensor_y & 0xffffu) << 16));      // td0[31:16]|td1[15:0]
    g1[3] = (int)((((unsigned)tensor_y >> 16) & 0xffffu) |
                  (((unsigned)tile_x & 0xffffu) << 16));        // td1[31:16]|tile_dim0
    g1[4] = (int)((unsigned)tile_y & 0xffffu);                  // tile_dim1 (tile_dim2=0)
    g1[5] = stride_x;                                           // tensor_dim0_stride[31:0]
    g1[6] = 0;
    g1[7] = 0;
    v4i gz4 = {0, 0, 0, 0};
    v8i gz8 = {0, 0, 0, 0, 0, 0, 0, 0};
    __builtin_amdgcn_tensor_load_to_lds(g0, g1, gz4, gz4, gz8, 0);
}

// ---------------- elementwise f32 -> bf16 (weight pre-conversion) ----------------
__global__ void cvt_f32_bf16_kernel(const float* __restrict__ src,
                                    __bf16* __restrict__ dst, int n) {
    int i = blockIdx.x * 256 + threadIdx.x;
    if (i < n) dst[i] = (__bf16)src[i];
}

// ---------------- pos2slot inverse map ----------------
__global__ void init_map_kernel(int* __restrict__ pos2slot) {
    int i = blockIdx.x * 256 + threadIdx.x;
    if (i < B_ * L_) pos2slot[i] = -1;
}

__global__ void scatter_map_kernel(const int* __restrict__ indices,
                                   int* __restrict__ pos2slot) {
    int i = blockIdx.x * 256 + threadIdx.x;
    if (i < B_ * K_) {
        int b = i / K_;
        pos2slot[b * L_ + indices[i]] = i % K_;
    }
}

// ---------------- fc1: h = x @ w1^T + b1 ----------------
// Block tile 128M x 64N, 8 waves, each wave 16M x 64N (4 accumulators).
// Pre-converted bf16 w1 panel (64 rows x 32 k) double-buffered in LDS via TDM.
__global__ __launch_bounds__(256)
void fc1_wmma_kernel(const float* __restrict__ x,      // [B,K,C] f32
                     const __bf16* __restrict__ w1b,   // [HID,C] bf16
                     const float* __restrict__ b1,     // [HID]
                     __bf16* __restrict__ hc)          // [B,K,HID] bf16
{
    __shared__ __bf16 bpanel[2][64 * 32];               // 2 x 4 KB
    const int nMb = K_ / 128;    // 32
    const int nNb = HID_ / 64;   // 24
    int b   = blockIdx.x / (nMb * nNb);
    int rem = blockIdx.x % (nMb * nNb);
    int mb  = rem / nNb;
    int nb  = rem % nNb;

    int wave = threadIdx.x >> 5;
    int lane = threadIdx.x & 31;
    int half = lane >> 4;
    int lidx = lane & 15;

    int m0 = mb * 128 + wave * 16;
    int n0 = nb * 64;
    const float* arow = x + ((size_t)b * K_ + m0 + lidx) * C_;

    unsigned lds0 = (unsigned)(uintptr_t)&bpanel[0][0];
    unsigned lds1 = (unsigned)(uintptr_t)&bpanel[1][0];

    const int NK = C_ / 32;      // 12
    if (wave == 0) {
        tdm_load_tile(lds0, w1b + (size_t)n0 * C_, 1, 32, 64, C_, HID_, C_);
        __builtin_amdgcn_s_wait_tensorcnt(0);
    }
    __syncthreads();

    v8f acc[4] = {{}, {}, {}, {}};
    for (int kc = 0; kc < NK; ++kc) {
        int k0 = kc * 32;
        const __bf16* bp = &bpanel[kc & 1][0];
        if (wave == 0 && kc + 1 < NK)
            tdm_load_tile((kc & 1) ? lds0 : lds1,
                          w1b + (size_t)n0 * C_ + (k0 + 32), 1, 32, 64, C_, HID_, C_);
        if (kc + 1 < NK) __builtin_prefetch(arow + k0 + 32, 0, 1);

        // A fragment: 16 f32 -> bf16 (reused by 4 WMMAs)
        v16bf af;
#pragma unroll
        for (int v = 0; v < 8; ++v) {
            int ka = k0 + kbaseA(v, half);
            af[2 * v]     = (__bf16)arow[ka];
            af[2 * v + 1] = (__bf16)arow[ka + 1];
        }
        // 4 B fragments: in the same contiguous k order as A, each lane's fragment is the 16
        // contiguous bf16 at row n_local, k = half*16 .. half*16+15.
#pragma unroll
        for (int j = 0; j < 4; ++j) {
            int nl = j * 16 + lidx;
            v16bf bfr = *(const v16bf*)(bp + nl * 32 + half * 16);
            acc[j] = __builtin_amdgcn_wmma_f32_16x16x32_bf16(
                false, af, false, bfr, (short)0, acc[j], false, false);
        }
        __syncthreads();                       // all waves done reading bp
        if (kc + 1 < NK) {
            if (wave == 0) __builtin_amdgcn_s_wait_tensorcnt(0);
            __syncthreads();                   // next panel visible
        }
    }

    __bf16* obase = hc + (size_t)b * K_ * HID_;
#pragma unroll
    for (int j = 0; j < 4; ++j) {
        int col = n0 + j * 16 + lidx;
        float bias = b1[col];
#pragma unroll
        for (int v = 0; v < 8; ++v) {
            int rr = m0 + v + half * 8;        // C/D layout: M = v + 8*half
            obase[(size_t)rr * HID_ + col] = (__bf16)(acc[j][v] + bias);
        }
    }
}

// ---------------- depthwise 3x3 conv on sparse grid + exact GELU ----------------
__global__ __launch_bounds__(256)
void conv_gelu_kernel(const __bf16* __restrict__ hc,        // [B,K,HID]
                      const int* __restrict__ indices,      // [B,K]
                      const int* __restrict__ pos2slot,     // [B,L]
                      const float* __restrict__ dww,        // [HID,1,3,3]
                      const float* __restrict__ dwb,        // [HID]
                      __bf16* __restrict__ act)             // [B,K,HID]
{
    int b    = blockIdx.x / K_;
    int slot = blockIdx.x % K_;

    __shared__ int nslot[9];
    if (threadIdx.x < 9) {
        int p  = indices[b * K_ + slot];
        int t  = p / (H_ * W_);
        int r  = p % (H_ * W_);
        int y  = r / W_;
        int xx = r % W_;
        int j  = threadIdx.x;
        int dy = j / 3 - 1, dx = j % 3 - 1;
        int ny = y + dy, nx = xx + dx;
        int s = -1;
        if (ny >= 0 && ny < H_ && nx >= 0 && nx < W_)
            s = pos2slot[b * L_ + t * H_ * W_ + ny * W_ + nx];
        nslot[j] = s;
    }
    __syncthreads();

    const __bf16* base   = hc  + (size_t)b * K_ * HID_;
    __bf16*       outrow = act + ((size_t)b * K_ + slot) * HID_;

    for (int h = threadIdx.x; h < HID_; h += 256) {
        float a = dwb[h];
#pragma unroll
        for (int j = 0; j < 9; ++j) {
            int s = nslot[j];
            if (s >= 0)
                a += (float)base[(size_t)s * HID_ + h] * dww[h * 9 + j];
        }
        float g = 0.5f * a * (1.0f + erff(a * 0.70710678118654752f));
        outrow[h] = (__bf16)g;
    }
}

// ---------------- fc2: out = act @ w2^T + b2 ----------------
__global__ __launch_bounds__(256)
void fc2_wmma_kernel(const __bf16* __restrict__ act,   // [B,K,HID] bf16
                     const __bf16* __restrict__ w2b,   // [C,HID] bf16
                     const float* __restrict__ b2,     // [C]
                     float* __restrict__ out)          // [B,K,C]
{
    __shared__ __bf16 bpanel[2][64 * 32];
    const int nMb = K_ / 128;   // 32
    const int nNb = C_ / 64;    // 6
    int b   = blockIdx.x / (nMb * nNb);
    int rem = blockIdx.x % (nMb * nNb);
    int mb  = rem / nNb;
    int nb  = rem % nNb;

    int wave = threadIdx.x >> 5;
    int lane = threadIdx.x & 31;
    int half = lane >> 4;
    int lidx = lane & 15;

    int m0 = mb * 128 + wave * 16;
    int n0 = nb * 64;
    const __bf16* arow = act + ((size_t)b * K_ + m0 + lidx) * HID_;

    unsigned lds0 = (unsigned)(uintptr_t)&bpanel[0][0];
    unsigned lds1 = (unsigned)(uintptr_t)&bpanel[1][0];

    const int NK = HID_ / 32;   // 48
    if (wave == 0) {
        tdm_load_tile(lds0, w2b + (size_t)n0 * HID_, 1, 32, 64, HID_, C_, HID_);
        __builtin_amdgcn_s_wait_tensorcnt(0);
    }
    __syncthreads();

    v8f acc[4] = {{}, {}, {}, {}};
    for (int kc = 0; kc < NK; ++kc) {
        int k0 = kc * 32;
        const __bf16* bp = &bpanel[kc & 1][0];
        if (wave == 0 && kc + 1 < NK)
            tdm_load_tile((kc & 1) ? lds0 : lds1,
                          w2b + (size_t)n0 * HID_ + (k0 + 32), 1, 32, 64, HID_, C_, HID_);
        if (kc + 1 < NK) __builtin_prefetch(arow + k0 + 32, 0, 1);

        // A fragment (bf16): contiguous 16 elems, two 8-elem chunks
        // at k0 + 16*half and k0 + 16*half + 8.
        v8bf alo = *(const v8bf*)(arow + k0 + 16 * half);
        v8bf ahi = *(const v8bf*)(arow + k0 + 16 * half + 8);
        v16bf af = __builtin_shufflevector(alo, ahi,
                                           0, 1, 2, 3, 4, 5, 6, 7,
                                           8, 9, 10, 11, 12, 13, 14, 15);
#pragma unroll
        for (int j = 0; j < 4; ++j) {
            int nl = j * 16 + lidx;
            v16bf bfr = *(const v16bf*)(bp + nl * 32 + half * 16);
            acc[j] = __builtin_amdgcn_wmma_f32_16x16x32_bf16(
                false, af, false, bfr, (short)0, acc[j], false, false);
        }
        __syncthreads();
        if (kc + 1 < NK) {
            if (wave == 0) __builtin_amdgcn_s_wait_tensorcnt(0);
            __syncthreads();
        }
    }

    float* obase = out + (size_t)b * K_ * C_;
#pragma unroll
    for (int j = 0; j < 4; ++j) {
        int col = n0 + j * 16 + lidx;
        float bias = b2[col];
#pragma unroll
        for (int v = 0; v < 8; ++v) {
            int rr = m0 + v + half * 8;
            obase[(size_t)rr * C_ + col] = acc[j][v] + bias;
        }
    }
}

// ---------------- launch ----------------
extern "C" void kernel_launch(void* const* d_in, const int* in_sizes, int n_in,
                              void* d_out, int out_size, void* d_ws, size_t ws_size,
                              hipStream_t stream) {
    // setup_inputs order: x_sparse, indices, B, L, H, W, w1, b1, dw_w, dw_b, w2, b2
    const float* x       = (const float*)d_in[0];
    const int*   indices = (const int*)d_in[1];
    const float* w1      = (const float*)d_in[6];
    const float* b1      = (const float*)d_in[7];
    const float* dww     = (const float*)d_in[8];
    const float* dwb     = (const float*)d_in[9];
    const float* w2      = (const float*)d_in[10];
    const float* b2      = (const float*)d_in[11];
    float*       out     = (float*)d_out;

    // workspace layout (bytes)
    char* ws = (char*)d_ws;
    int*    pos2slot = (int*)ws;                               // 131072
    __bf16* hc       = (__bf16*)(ws + 131072);                 // 50331648
    __bf16* act      = (__bf16*)(ws + 50462720);               // 50331648
    __bf16* w1b      = (__bf16*)(ws + 100794368);              // 1179648
    __bf16* w2b      = (__bf16*)(ws + 101974016);              // 1179648

    const int NW = HID_ * C_;  // 589824 weights each

    // 0) one-time weight pre-conversion to bf16
    cvt_f32_bf16_kernel<<<(NW + 255) / 256, 256, 0, stream>>>(w1, w1b, NW);
    cvt_f32_bf16_kernel<<<(NW + 255) / 256, 256, 0, stream>>>(w2, w2b, NW);

    // 1) inverse map
    init_map_kernel<<<(B_ * L_ + 255) / 256, 256, 0, stream>>>(pos2slot);
    scatter_map_kernel<<<(B_ * K_ + 255) / 256, 256, 0, stream>>>(indices, pos2slot);

    // 2) fc1: (4096/128)*(1536/64)*4 = 3072 blocks
    fc1_wmma_kernel<<<(K_ / 128) * (HID_ / 64) * B_, 256, 0, stream>>>(x, w1b, b1, hc);

    // 3) depthwise conv + GELU
    conv_gelu_kernel<<<B_ * K_, 256, 0, stream>>>(hc, indices, pos2slot, dww, dwb, act);

    // 4) fc2: (4096/128)*(384/64)*4 = 768 blocks
    fc2_wmma_kernel<<<(K_ / 128) * (C_ / 64) * B_, 256, 0, stream>>>(act, w2b, b2, out);
}
